// QLSTMBlock_10247791968829
// MI455X (gfx1250) — compile-verified, matrix operand order checked by reading, not run on hardware
//
#include <hip/hip_runtime.h>
#include <cstdint>
#include <cstddef>

// ---------------------------------------------------------------------------
// QLSTM block for MI455X (gfx1250, wave32, WMMA).
// B=4, L=4096, D=1024, F=4096.  All GEMMs run on v_wmma_f32_16x16x32_bf16
// with double-buffered async global->LDS staging (one barrier per K-step).
// ---------------------------------------------------------------------------

typedef __bf16 bf16_t;
typedef __attribute__((ext_vector_type(8)))  __bf16 v8bf;
typedef __attribute__((ext_vector_type(16))) __bf16 v16bf;
typedef __attribute__((ext_vector_type(8)))  float  v8f;
typedef __attribute__((ext_vector_type(4)))  int    i32x4;

#define DEV __device__ __forceinline__

DEV float rcp_(float x) { return __builtin_amdgcn_rcpf(x); }
DEV float sigmoid_(float x) { return rcp_(1.0f + __expf(-x)); }
DEV float tanh_(float x) { return 2.0f * sigmoid_(2.0f * x) - 1.0f; }

// ---- async global->LDS copy (gfx1250 GLOBAL_LOAD_ASYNC_TO_LDS_B128), with
// ---- a synchronous load+ds_store fallback if the builtin is unavailable.
DEV void cp_b128(void* lds, const void* g) {
#if __has_builtin(__builtin_amdgcn_global_load_async_to_lds_b128)
  __builtin_amdgcn_global_load_async_to_lds_b128(
      (__attribute__((address_space(1))) i32x4*)g,
      (__attribute__((address_space(3))) i32x4*)lds, 0, 0);
#else
  *(v8bf*)lds = *(const v8bf*)g;
#endif
}
DEV void cp_wait() {
#if __has_builtin(__builtin_amdgcn_global_load_async_to_lds_b128)
  asm volatile("s_wait_asynccnt 0" ::: "memory");
#endif
}

// Build a 16x32 bf16 A/B fragment row from LDS (row stride 32 elements).
// Lane l (<16): K {0..7, 16..23}; lane l (>=16): K {8..15, 24..31}.
DEV v16bf make_frag(const bf16_t* rowp, int k0) {
  const v8bf lo = *(const v8bf*)(rowp + k0);
  const v8bf hi = *(const v8bf*)(rowp + k0 + 16);
  v16bf r;
#pragma unroll
  for (int i = 0; i < 8; ++i) { r[i] = lo[i]; r[i + 8] = hi[i]; }
  return r;
}

DEV v8f wmma_bf16(v16bf a, v16bf b, v8f c) {
  return __builtin_amdgcn_wmma_f32_16x16x32_bf16(
      false, a, false, b, (short)0, c, false, false);
}

// ---------------------------------------------------------------------------
// Weight fp32 [R][C] -> bf16 transposed [C][R]   (makes B K-contiguous)
// ---------------------------------------------------------------------------
__global__ __launch_bounds__(256) void transpose_to_bf16(
    const float* __restrict__ W, bf16_t* __restrict__ WT, int R, int C) {
  __shared__ float tile[32][33];
  const int cb = blockIdx.x * 32, rb = blockIdx.y * 32;
  for (int i = threadIdx.y; i < 32; i += 8)
    tile[i][threadIdx.x] = W[(size_t)(rb + i) * C + cb + threadIdx.x];
  __syncthreads();
  for (int i = threadIdx.y; i < 32; i += 8)
    WT[(size_t)(cb + i) * R + rb + threadIdx.x] = (bf16_t)tile[threadIdx.x][i];
}

// ---------------------------------------------------------------------------
// RMSNorm row (D=1024) -> bf16
// ---------------------------------------------------------------------------
__global__ __launch_bounds__(256) void rmsnorm_kernel(
    const float* __restrict__ x, const float* __restrict__ g,
    bf16_t* __restrict__ out) {
  const int D = 1024;
  const size_t row = blockIdx.x;
  const float* xr = x + row * D;
  float v[4];
  float ss = 0.f;
#pragma unroll
  for (int j = 0; j < 4; ++j) {
    v[j] = xr[threadIdx.x + j * 256];
    ss += v[j] * v[j];
  }
#pragma unroll
  for (int off = 16; off > 0; off >>= 1) ss += __shfl_down(ss, off);
  __shared__ float red[8];
  if ((threadIdx.x & 31) == 0) red[threadIdx.x >> 5] = ss;
  __syncthreads();
  const float tot = red[0] + red[1] + red[2] + red[3] +
                    red[4] + red[5] + red[6] + red[7];
  const float rstd = rsqrtf(tot * (1.0f / D) + 1e-6f);
  bf16_t* orow = out + row * D;
#pragma unroll
  for (int j = 0; j < 4; ++j) {
    const int i = threadIdx.x + j * 256;
    orow[i] = (bf16_t)(v[j] * rstd * g[i]);
  }
}

// ---------------------------------------------------------------------------
// Gates GEMM: A[M,K]bf16 x {Wf,Wi,Wig,Wog}^T[N,K]bf16 -> forget/inp/og f32
// Block tile 32x64, 8 waves (2x4) of 16x16 WMMA tiles, K-step 32,
// double-buffered LDS, one barrier per step.
// ---------------------------------------------------------------------------
__global__ __launch_bounds__(256) void gemm_gates_kernel(
    const bf16_t* __restrict__ A,
    const bf16_t* __restrict__ B0, const bf16_t* __restrict__ B1,
    const bf16_t* __restrict__ B2, const bf16_t* __restrict__ B3,
    const float* __restrict__ bf_, const float* __restrict__ bi_,
    const float* __restrict__ big_, const float* __restrict__ bog_,
    float* __restrict__ Fo, float* __restrict__ Io, float* __restrict__ Oo,
    int M, int N, int K) {
  __shared__ __align__(16) bf16_t As[2][32][32];
  __shared__ __align__(16) bf16_t Bs[2][4][64][32];
  const int bm = blockIdx.x * 32;
  const int bn = blockIdx.y * 64;
  const int lane = threadIdx.x & 31;
  const int wave = threadIdx.x >> 5;
  const int wm = (wave & 1) * 16;
  const int wn = (wave >> 1) * 16;
  const bf16_t* Bp[4] = {B0, B1, B2, B3};
  const int r = threadIdx.x >> 2;
  const int cc = (threadIdx.x & 3) * 8;
  v8f acc0 = {}, acc1 = {}, acc2 = {}, acc3 = {};

  auto stage = [&](int step) {
    const int k0 = step * 32;
    const int buf = step & 1;
    if (threadIdx.x < 128)
      cp_b128(&As[buf][r][cc], A + (size_t)(bm + r) * K + k0 + cc);
#pragma unroll
    for (int g = 0; g < 4; ++g)
      cp_b128(&Bs[buf][g][r][cc], Bp[g] + (size_t)(bn + r) * K + k0 + cc);
  };

  const int nsteps = K >> 5;
  stage(0);
  for (int i = 0; i < nsteps; ++i) {
    cp_wait();
    __syncthreads();
    if (i + 1 < nsteps) stage(i + 1);
    const int buf = i & 1;
    const int kk = (lane < 16) ? 0 : 8;
    const int lr = lane & 15;
    const v16bf af  = make_frag(&As[buf][wm + lr][0], kk);
    const v16bf fb0 = make_frag(&Bs[buf][0][wn + lr][0], kk);
    const v16bf fb1 = make_frag(&Bs[buf][1][wn + lr][0], kk);
    const v16bf fb2 = make_frag(&Bs[buf][2][wn + lr][0], kk);
    const v16bf fb3 = make_frag(&Bs[buf][3][wn + lr][0], kk);
    acc0 = wmma_bf16(af, fb0, acc0);
    acc1 = wmma_bf16(af, fb1, acc1);
    acc2 = wmma_bf16(af, fb2, acc2);
    acc3 = wmma_bf16(af, fb3, acc3);
  }
  const int n = bn + wn + (lane & 15);
  const int mb = bm + wm + ((lane < 16) ? 0 : 8);
  const float vbf = bf_[n], vbi = bi_[n], vbig = big_[n], vbog = bog_[n];
#pragma unroll
  for (int v = 0; v < 8; ++v) {
    const size_t idx = (size_t)(mb + v) * N + n;
    Fo[idx] = sigmoid_(acc0[v] + vbf);
    Io[idx] = tanh_(acc1[v] + vbi) * sigmoid_(acc2[v] + vbig);
    Oo[idx] = sigmoid_(acc3[v] + vbog);
  }
}

// ---------------------------------------------------------------------------
// SwiGLU GEMM: xn2[M,K] x {Wfc,Wact}^T[N,K] -> ff[M,N] bf16
// Block tile 64x128, 8 waves (2x4), each wave a 32x32 macro-tile per matrix.
// ---------------------------------------------------------------------------
__global__ __launch_bounds__(256) void gemm_swiglu_kernel(
    const bf16_t* __restrict__ A,
    const bf16_t* __restrict__ B0, const bf16_t* __restrict__ B1,
    const float* __restrict__ bfc, const float* __restrict__ bact,
    bf16_t* __restrict__ ff, int M, int N, int K) {
  __shared__ __align__(16) bf16_t As[2][64][32];
  __shared__ __align__(16) bf16_t Bs[2][2][128][32];
  const int bm = blockIdx.x * 64;
  const int bn = blockIdx.y * 128;
  const int lane = threadIdx.x & 31;
  const int wave = threadIdx.x >> 5;
  const int wm = (wave & 1) * 32;
  const int wn = (wave >> 1) * 32;
  const bf16_t* Bp[2] = {B0, B1};
  const int r = threadIdx.x >> 2;
  const int cc = (threadIdx.x & 3) * 8;
  v8f f00 = {}, f01 = {}, f10 = {}, f11 = {};
  v8f g00 = {}, g01 = {}, g10 = {}, g11 = {};

  auto stage = [&](int step) {
    const int k0 = step * 32;
    const int buf = step & 1;
    cp_b128(&As[buf][r][cc], A + (size_t)(bm + r) * K + k0 + cc);
#pragma unroll
    for (int g = 0; g < 2; ++g) {
      cp_b128(&Bs[buf][g][r][cc], Bp[g] + (size_t)(bn + r) * K + k0 + cc);
      cp_b128(&Bs[buf][g][r + 64][cc],
              Bp[g] + (size_t)(bn + r + 64) * K + k0 + cc);
    }
  };

  const int nsteps = K >> 5;
  stage(0);
  for (int i = 0; i < nsteps; ++i) {
    cp_wait();
    __syncthreads();
    if (i + 1 < nsteps) stage(i + 1);
    const int buf = i & 1;
    const int kk = (lane < 16) ? 0 : 8;
    const int lr = lane & 15;
    const v16bf a0 = make_frag(&As[buf][wm + lr][0], kk);
    const v16bf a1 = make_frag(&As[buf][wm + 16 + lr][0], kk);
    const v16bf p0 = make_frag(&Bs[buf][0][wn + lr][0], kk);
    const v16bf p1 = make_frag(&Bs[buf][0][wn + 16 + lr][0], kk);
    const v16bf q0 = make_frag(&Bs[buf][1][wn + lr][0], kk);
    const v16bf q1 = make_frag(&Bs[buf][1][wn + 16 + lr][0], kk);
    f00 = wmma_bf16(a0, p0, f00);
    f01 = wmma_bf16(a0, p1, f01);
    f10 = wmma_bf16(a1, p0, f10);
    f11 = wmma_bf16(a1, p1, f11);
    g00 = wmma_bf16(a0, q0, g00);
    g01 = wmma_bf16(a0, q1, g01);
    g10 = wmma_bf16(a1, q0, g10);
    g11 = wmma_bf16(a1, q1, g11);
  }
  const int lr = lane & 15;
  const int mh = (lane < 16) ? 0 : 8;
  const int n0 = bn + wn + lr;
  const int n1 = n0 + 16;
  const float bfc0 = bfc[n0], bfc1 = bfc[n1];
  const float bac0 = bact[n0], bac1 = bact[n1];
  const int m0 = bm + wm + mh;
  const int m1 = m0 + 16;
#pragma unroll
  for (int v = 0; v < 8; ++v) {
    const size_t i0 = (size_t)(m0 + v) * N + n0;
    const size_t i1 = (size_t)(m1 + v) * N + n0;
    float a;
    a = g00[v] + bac0; ff[i0]      = (bf16_t)((f00[v] + bfc0) * a * sigmoid_(a));
    a = g01[v] + bac1; ff[i0 + 16] = (bf16_t)((f01[v] + bfc1) * a * sigmoid_(a));
    a = g10[v] + bac0; ff[i1]      = (bf16_t)((f10[v] + bfc0) * a * sigmoid_(a));
    a = g11[v] + bac1; ff[i1 + 16] = (bf16_t)((f11[v] + bfc1) * a * sigmoid_(a));
  }
}

// ---------------------------------------------------------------------------
// Output GEMM with residual: out[M,N] = x2[M,N] + ff[M,K] x Wout^T[N,K] + b
// Block tile 64x128, each wave a 32x32 macro-tile.
// ---------------------------------------------------------------------------
__global__ __launch_bounds__(256) void gemm_out_kernel(
    const bf16_t* __restrict__ A, const bf16_t* __restrict__ B0,
    const float* __restrict__ bout, const float* __restrict__ x2,
    float* __restrict__ out, int M, int N, int K) {
  __shared__ __align__(16) bf16_t As[2][64][32];
  __shared__ __align__(16) bf16_t Bs[2][128][32];
  const int bm = blockIdx.x * 64;
  const int bn = blockIdx.y * 128;
  const int lane = threadIdx.x & 31;
  const int wave = threadIdx.x >> 5;
  const int wm = (wave & 1) * 32;
  const int wn = (wave >> 1) * 32;
  const int r = threadIdx.x >> 2;
  const int cc = (threadIdx.x & 3) * 8;
  v8f c00 = {}, c01 = {}, c10 = {}, c11 = {};

  auto stage = [&](int step) {
    const int k0 = step * 32;
    const int buf = step & 1;
    cp_b128(&As[buf][r][cc], A + (size_t)(bm + r) * K + k0 + cc);
    cp_b128(&Bs[buf][r][cc], B0 + (size_t)(bn + r) * K + k0 + cc);
    cp_b128(&Bs[buf][r + 64][cc], B0 + (size_t)(bn + r + 64) * K + k0 + cc);
  };

  const int nsteps = K >> 5;
  stage(0);
  for (int i = 0; i < nsteps; ++i) {
    cp_wait();
    __syncthreads();
    if (i + 1 < nsteps) stage(i + 1);
    const int buf = i & 1;
    const int kk = (lane < 16) ? 0 : 8;
    const int lr = lane & 15;
    const v16bf a0 = make_frag(&As[buf][wm + lr][0], kk);
    const v16bf a1 = make_frag(&As[buf][wm + 16 + lr][0], kk);
    const v16bf b0 = make_frag(&Bs[buf][wn + lr][0], kk);
    const v16bf b1 = make_frag(&Bs[buf][wn + 16 + lr][0], kk);
    c00 = wmma_bf16(a0, b0, c00);
    c01 = wmma_bf16(a0, b1, c01);
    c10 = wmma_bf16(a1, b0, c10);
    c11 = wmma_bf16(a1, b1, c11);
  }
  const int lr = lane & 15;
  const int mh = (lane < 16) ? 0 : 8;
  const int n0 = bn + wn + lr;
  const float vb0 = bout[n0], vb1 = bout[n0 + 16];
  const int m0 = bm + wm + mh;
  const int m1 = m0 + 16;
#pragma unroll
  for (int v = 0; v < 8; ++v) {
    const size_t i0 = (size_t)(m0 + v) * N + n0;
    const size_t i1 = (size_t)(m1 + v) * N + n0;
    out[i0]      = x2[i0]      + c00[v] + vb0;
    out[i0 + 16] = x2[i0 + 16] + c01[v] + vb1;
    out[i1]      = x2[i1]      + c10[v] + vb0;
    out[i1 + 16] = x2[i1 + 16] + c11[v] + vb1;
  }
}

// ---------------------------------------------------------------------------
// Chunked linear-recurrence scan: h_t = f_t * h_{t-1} + i_t, h_0 = hidden.
// C=32 chunks of Lc=128 -> 131072-way parallel phases + tiny serial carry.
// ---------------------------------------------------------------------------
__global__ __launch_bounds__(256) void scan_reduce_kernel(
    const float* __restrict__ Fo, const float* __restrict__ Io,
    float* __restrict__ aprod, float* __restrict__ bend) {
  const int D = 1024, L = 4096, Lc = 128;
  const int tid = blockIdx.x * 256 + threadIdx.x;     // [b][c][d]
  const int d = tid & (D - 1);
  const int c = (tid >> 10) & 31;
  const int b = tid >> 15;
  size_t pos = ((size_t)b * L + (size_t)c * Lc) * D + d;
  float a = 1.f, s = 0.f;
  for (int t = 0; t < Lc; ++t, pos += D) {
    const float f = Fo[pos];
    a *= f;
    s = f * s + Io[pos];
  }
  aprod[tid] = a;
  bend[tid] = s;
}

__global__ __launch_bounds__(256) void scan_carry_kernel(
    const float* __restrict__ hidden, const float* __restrict__ aprod,
    const float* __restrict__ bend, float* __restrict__ carry,
    float* __restrict__ hlast) {
  const int D = 1024, C = 32;
  const int tid = blockIdx.x * 256 + threadIdx.x;     // [b][d]
  const int d = tid & (D - 1);
  const int b = tid >> 10;
  float h = hidden[tid];
  for (int c = 0; c < C; ++c) {
    const size_t i = ((size_t)b * C + c) * D + d;
    carry[i] = h;
    h = aprod[i] * h + bend[i];
  }
  hlast[tid] = h;
}

__global__ __launch_bounds__(256) void scan_apply_kernel(
    const float* __restrict__ Fo, const float* __restrict__ Io,
    const float* __restrict__ Og, const float* __restrict__ xin,
    const float* __restrict__ carry, float* __restrict__ x2) {
  const int D = 1024, L = 4096, Lc = 128;
  const int tid = blockIdx.x * 256 + threadIdx.x;     // [b][c][d]
  const int d = tid & (D - 1);
  const int c = (tid >> 10) & 31;
  const int b = tid >> 15;
  size_t pos = ((size_t)b * L + (size_t)c * Lc) * D + d;
  float h = carry[tid];
  for (int t = 0; t < Lc; ++t, pos += D) {
    h = Fo[pos] * h + Io[pos];
    x2[pos] = xin[pos] + tanh_(h) * Og[pos];
  }
}

// ---------------------------------------------------------------------------
extern "C" void kernel_launch(void* const* d_in, const int* in_sizes, int n_in,
                              void* d_out, int out_size, void* d_ws,
                              size_t ws_size, hipStream_t stream) {
  (void)in_sizes; (void)n_in; (void)out_size; (void)ws_size;
  const int B = 4, L = 4096, D = 1024, F = 4096;
  const int M = B * L;          // 16384
  const int C = 32;             // scan chunks

  const float* x      = (const float*)d_in[0];
  const float* hidden = (const float*)d_in[1];
  const float* g1     = (const float*)d_in[2];
  const float* g2     = (const float*)d_in[3];
  const float* W_f    = (const float*)d_in[4];
  const float* b_f    = (const float*)d_in[5];
  const float* W_i    = (const float*)d_in[6];
  const float* b_i    = (const float*)d_in[7];
  const float* W_ig   = (const float*)d_in[8];
  const float* b_ig   = (const float*)d_in[9];
  const float* W_og   = (const float*)d_in[10];
  const float* b_og   = (const float*)d_in[11];
  const float* W_fc   = (const float*)d_in[12];
  const float* b_fc   = (const float*)d_in[13];
  const float* W_act  = (const float*)d_in[14];
  const float* b_act  = (const float*)d_in[15];
  const float* W_out  = (const float*)d_in[16];
  const float* b_out  = (const float*)d_in[17];

  char* ws = (char*)d_ws;
  auto take = [&](size_t bytes) {
    void* p = (void*)ws;
    ws += (bytes + 255) & ~(size_t)255;
    return p;
  };
  bf16_t* xn    = (bf16_t*)take((size_t)M * D * 2);
  bf16_t* xn2   = (bf16_t*)take((size_t)M * D * 2);
  bf16_t* WfT   = (bf16_t*)take((size_t)D * D * 2);
  bf16_t* WiT   = (bf16_t*)take((size_t)D * D * 2);
  bf16_t* WigT  = (bf16_t*)take((size_t)D * D * 2);
  bf16_t* WogT  = (bf16_t*)take((size_t)D * D * 2);
  bf16_t* WfcT  = (bf16_t*)take((size_t)F * D * 2);
  bf16_t* WactT = (bf16_t*)take((size_t)F * D * 2);
  bf16_t* WoutT = (bf16_t*)take((size_t)D * F * 2);
  float*  Fo    = (float*)take((size_t)M * D * 4);
  float*  Io    = (float*)take((size_t)M * D * 4);
  float*  Og    = (float*)take((size_t)M * D * 4);
  bf16_t* ff    = (bf16_t*)take((size_t)M * F * 2);
  float*  aprod = (float*)take((size_t)B * C * D * 4);
  float*  bend  = (float*)take((size_t)B * C * D * 4);
  float*  carry = (float*)take((size_t)B * C * D * 4);

  float* xout  = (float*)d_out;                 // [B,L,D]
  float* hlast = (float*)d_out + (size_t)M * D; // [B,D]

  const dim3 tb(32, 8);
  // Weight transposes (fp32 -> bf16, K-contiguous rows)
  transpose_to_bf16<<<dim3(D / 32, D / 32), tb, 0, stream>>>(W_f,  WfT,  D, D);
  transpose_to_bf16<<<dim3(D / 32, D / 32), tb, 0, stream>>>(W_i,  WiT,  D, D);
  transpose_to_bf16<<<dim3(D / 32, D / 32), tb, 0, stream>>>(W_ig, WigT, D, D);
  transpose_to_bf16<<<dim3(D / 32, D / 32), tb, 0, stream>>>(W_og, WogT, D, D);
  transpose_to_bf16<<<dim3(F / 32, D / 32), tb, 0, stream>>>(W_fc,  WfcT,  D, F);
  transpose_to_bf16<<<dim3(F / 32, D / 32), tb, 0, stream>>>(W_act, WactT, D, F);
  transpose_to_bf16<<<dim3(D / 32, F / 32), tb, 0, stream>>>(W_out, WoutT, F, D);

  // QLSTM branch
  rmsnorm_kernel<<<M, 256, 0, stream>>>(x, g1, xn);
  gemm_gates_kernel<<<dim3(M / 32, D / 64), 256, 0, stream>>>(
      xn, WfT, WiT, WigT, WogT, b_f, b_i, b_ig, b_og, Fo, Io, Og, M, D, D);
  scan_reduce_kernel<<<(B * C * D) / 256, 256, 0, stream>>>(Fo, Io, aprod, bend);
  scan_carry_kernel<<<(B * D) / 256, 256, 0, stream>>>(hidden, aprod, bend,
                                                       carry, hlast);
  scan_apply_kernel<<<(B * C * D) / 256, 256, 0, stream>>>(Fo, Io, Og, x,
                                                           carry, xout);
  // FFN branch (x2 lives in d_out; final GEMM adds residual in place)
  rmsnorm_kernel<<<M, 256, 0, stream>>>(xout, g2, xn2);
  gemm_swiglu_kernel<<<dim3(M / 64, F / 128), 256, 0, stream>>>(
      xn2, WfcT, WactT, b_fc, b_act, ff, M, F, D);
  gemm_out_kernel<<<dim3(M / 64, D / 128), 256, 0, stream>>>(
      ff, WoutT, b_out, xout, xout, M, D, F);
}
